// MultiAgentNetwork_81063212745124
// MI455X (gfx1250) — compile-verified, matrix operand order checked by reading, not run on hardware
//
#include <hip/hip_runtime.h>
#include <hip/hip_bf16.h>
#include <math.h>

#define NP      4
#define HID     512
#define BATCH   16384
#define KENC    260
#define KPAD    288          // 9 * 32
#define KT1     9
#define KT2     16
#define NT_H    32           // 512 / 16
#define STR1    296          // padded LDS stride for layer-1 input
#define STR2    520          // padded LDS stride for hidden activations
#define MT      64           // batch rows per block
#define PI_F    3.14159265358979f

typedef _Float16 v16h __attribute__((ext_vector_type(16)));
typedef _Float16 v8h  __attribute__((ext_vector_type(8)));
typedef float    v8f  __attribute__((ext_vector_type(8)));

// K/N position of element i (0..15) of lane's A/B WMMA fragment (f16 16x16x32).
__device__ __forceinline__ void frag_kn(int lane, int i, int& kl, int& nl) {
  int v = i >> 1, r = i & 1;
  int kk = (v < 4) ? (2 * v) : (16 + 2 * (v - 4));
  kl = ((lane & 16) ? 8 : 0) + kk + r;
  nl = lane & 15;
}

// ---------------- encode: obs -> f16 features, K padded to 288 ----------------
__global__ void encode_kernel(const float* __restrict__ obs,
                              const float* __restrict__ freqs,
                              _Float16* __restrict__ enc) {
  int idx = blockIdx.x * blockDim.x + threadIdx.x;
  if (idx >= BATCH * KPAD) return;
  int b = idx / KPAD, j = idx - b * KPAD;
  float v = 0.f;
  if (j < 256) {
    int pi = j >> 6, jj = j & 63;
    float a = obs[b * 8 + pi];
    float f = freqs[jj & 31];
    v = (jj < 32) ? cosf(a * f) : sinf(a * f);
  } else if (j < KENC) {
    v = obs[b * 8 + 4 + (j - 256)];
  }
  enc[idx] = (_Float16)v;
}

// ---------------- routing: bucket sample ids by argmax(one_hot) ----------------
__global__ void zero_cnt_kernel(int* __restrict__ cnt) {
  if (threadIdx.x < NP) cnt[threadIdx.x] = 0;
}

__global__ void route_kernel(const float* __restrict__ obs,
                             int* __restrict__ cnt, int* __restrict__ bucket) {
  int b = blockIdx.x * blockDim.x + threadIdx.x;
  if (b >= BATCH) return;
  float best = obs[b * 8 + 4];
  int pid = 0;
  #pragma unroll
  for (int i = 1; i < NP; i++) {
    float v = obs[b * 8 + 4 + i];
    if (v > best) { best = v; pid = i; }   // strict > keeps first max (argmax)
  }
  int pos = atomicAdd(&cnt[pid], 1);
  bucket[pid * BATCH + pos] = b;
}

// -------- f32 weights [P][K][N] -> f16 in WMMA B-fragment order [nt][kt][lane][16]
__global__ void swizzle_w_kernel(const float* __restrict__ src,
                                 _Float16* __restrict__ dst,
                                 int K, int KT, int NT) {
  int p = blockIdx.y;
  int total = NT * KT * 512;                       // f16 elems per player
  int d = blockIdx.x * blockDim.x + threadIdx.x;
  if (d >= total) return;
  int i = d & 15;
  int lane = (d >> 4) & 31;
  int fi = d >> 9;                                 // nt*KT + kt
  int kt = fi % KT, nt = fi / KT;
  int kl, nl;
  frag_kn(lane, i, kl, nl);
  int k = kt * 32 + kl;
  int ncol = nt * 16 + nl;
  int N = NT * 16;
  float v = (k < K) ? src[((size_t)p * K + k) * N + ncol] : 0.f;
  dst[(size_t)p * total + d] = (_Float16)v;
}

// ---------------- fused per-player MLP over one 64-row tile ----------------
__device__ __forceinline__ void gemm_layer(const _Float16* hin, int strIn, int KT,
                                           _Float16* hout, int strOut,
                                           const _Float16* __restrict__ Wf,
                                           const float* __restrict__ bias,
                                           int lane, int mt, int nhalf) {
  v8f acc[16];
  #pragma unroll
  for (int i = 0; i < 16; i++) { v8f z = {}; acc[i] = z; }

  int m = mt * 16 + (lane & 15);
  int kofs = (lane & 16) ? 8 : 0;

  for (int kt = 0; kt < KT; kt++) {
    v16h a;
    int kb = kt * 32 + kofs;
    #pragma unroll
    for (int v = 0; v < 8; v++) {
      int kl = kb + ((v < 4) ? 2 * v : 16 + 2 * (v - 4));
      a[2 * v]     = hin[m * strIn + kl];
      a[2 * v + 1] = hin[m * strIn + kl + 1];
    }
    #pragma unroll
    for (int nt2 = 0; nt2 < 16; nt2++) {
      int nt = nhalf * 16 + nt2;
      const v8h* bp = (const v8h*)(Wf + (size_t)(nt * KT + kt) * 512 + lane * 16);
      v8h blo = bp[0];
      v8h bhi = bp[1];
      v16h b = __builtin_shufflevector(blo, bhi,
                                       0, 1, 2, 3, 4, 5, 6, 7,
                                       8, 9, 10, 11, 12, 13, 14, 15);
      acc[nt2] = __builtin_amdgcn_wmma_f32_16x16x32_f16(
          false, a, false, b, (short)0, acc[nt2], false, false);
    }
  }
  // epilogue: bias + ReLU, f16 back to LDS
  int mbase = mt * 16 + ((lane & 16) ? 8 : 0);
  #pragma unroll
  for (int nt2 = 0; nt2 < 16; nt2++) {
    int ncol = nhalf * 256 + nt2 * 16 + (lane & 15);
    float bv = bias[ncol];
    #pragma unroll
    for (int r = 0; r < 8; r++) {
      float x = acc[nt2][r] + bv;
      x = fmaxf(x, 0.f);
      hout[(mbase + r) * strOut + ncol] = (_Float16)x;
    }
  }
}

__global__ __launch_bounds__(256)
void mlp_kernel(const int* __restrict__ bucket, const int* __restrict__ cnt,
                const _Float16* __restrict__ enc,
                const _Float16* __restrict__ W1s, const _Float16* __restrict__ W2s,
                const _Float16* __restrict__ W3s,
                const float* __restrict__ b1, const float* __restrict__ b2,
                const float* __restrict__ b3,
                const float* __restrict__ Wd, const float* __restrict__ bd,
                float* __restrict__ out, int outOff) {
  __shared__ _Float16 buf0[MT * STR2];
  __shared__ _Float16 buf1[MT * STR2];
  __shared__ float part[MT][4];

  int p = blockIdx.y;
  int tile = blockIdx.x;
  int n = cnt[p];
  if (tile * MT >= n) return;

  int t = threadIdx.x;
  int lane = t & 31;
  int wave = t >> 5;
  int mt = wave & 3;
  int nhalf = wave >> 2;
  int base = tile * MT;

  // stage encoded rows (bucket-gather) into LDS; pad invalid rows with zeros
  for (int idx = t; idx < MT * KPAD; idx += 256) {
    int r = idx / KPAD, c = idx - r * KPAD;
    int g = base + r;
    _Float16 v = (_Float16)0.f;
    if (g < n) v = enc[(size_t)bucket[p * BATCH + g] * KPAD + c];
    buf0[r * STR1 + c] = v;
  }
  __syncthreads();

  const size_t w1p = (size_t)p * (NT_H * KT1 * 512);
  const size_t w2p = (size_t)p * (NT_H * KT2 * 512);
  gemm_layer(buf0, STR1, KT1, buf1, STR2, W1s + w1p, b1 + p * HID, lane, mt, nhalf);
  __syncthreads();
  gemm_layer(buf1, STR2, KT2, buf0, STR2, W2s + w2p, b2 + p * HID, lane, mt, nhalf);
  __syncthreads();
  gemm_layer(buf0, STR2, KT2, buf1, STR2, W3s + w2p, b3 + p * HID, lane, mt, nhalf);
  __syncthreads();

  // decoder: 512 -> 1, pi * tanh
  int r = t >> 2, q = t & 3;
  const _Float16* hrow = buf1 + r * STR2;
  float s = 0.f;
  #pragma unroll 4
  for (int k = q * 128; k < q * 128 + 128; k++)
    s += (float)hrow[k] * Wd[p * HID + k];
  part[r][q] = s;
  __syncthreads();
  if (q == 0) {
    float tot = part[r][0] + part[r][1] + part[r][2] + part[r][3] + bd[p];
    int g = base + r;
    if (g < n) out[outOff + bucket[p * BATCH + g]] = PI_F * tanhf(tot);
  }
}

// ---------------- host side ----------------
extern "C" void kernel_launch(void* const* d_in, const int* in_sizes, int n_in,
                              void* d_out, int out_size, void* d_ws, size_t ws_size,
                              hipStream_t stream) {
  const float* obs   = (const float*)d_in[0];
  const float* freqs = (const float*)d_in[1];
  const float* pW1 = (const float*)d_in[2];
  const float* pb1 = (const float*)d_in[3];
  const float* pW2 = (const float*)d_in[4];
  const float* pb2 = (const float*)d_in[5];
  const float* pW3 = (const float*)d_in[6];
  const float* pb3 = (const float*)d_in[7];
  const float* pWd = (const float*)d_in[8];
  const float* pbd = (const float*)d_in[9];
  const float* vW1 = (const float*)d_in[10];
  const float* vb1 = (const float*)d_in[11];
  const float* vW2 = (const float*)d_in[12];
  const float* vb2 = (const float*)d_in[13];
  const float* vW3 = (const float*)d_in[14];
  const float* vb3 = (const float*)d_in[15];
  const float* vWd = (const float*)d_in[16];
  const float* vbd = (const float*)d_in[17];
  float* out = (float*)d_out;

  // workspace carve-up
  const size_t W1TOT = (size_t)NP * NT_H * KT1 * 512;   // f16 elems
  const size_t W2TOT = (size_t)NP * NT_H * KT2 * 512;
  char* ws = (char*)d_ws;
  size_t off = 0;
  _Float16* enc = (_Float16*)(ws + off); off += (size_t)BATCH * KPAD * 2;
  int* cnt      = (int*)(ws + off);      off += 64;
  int* bucket   = (int*)(ws + off);      off += (size_t)NP * BATCH * 4;
  _Float16* pW1s = (_Float16*)(ws + off); off += W1TOT * 2;
  _Float16* pW2s = (_Float16*)(ws + off); off += W2TOT * 2;
  _Float16* pW3s = (_Float16*)(ws + off); off += W2TOT * 2;
  _Float16* vW1s = (_Float16*)(ws + off); off += W1TOT * 2;
  _Float16* vW2s = (_Float16*)(ws + off); off += W2TOT * 2;
  _Float16* vW3s = (_Float16*)(ws + off); off += W2TOT * 2;
  (void)off; (void)ws_size; (void)in_sizes; (void)n_in; (void)out_size;

  zero_cnt_kernel<<<1, 32, 0, stream>>>(cnt);
  encode_kernel<<<(BATCH * KPAD + 255) / 256, 256, 0, stream>>>(obs, freqs, enc);
  route_kernel<<<(BATCH + 255) / 256, 256, 0, stream>>>(obs, cnt, bucket);

  int per1 = NT_H * KT1 * 512, per2 = NT_H * KT2 * 512;
  dim3 g1((per1 + 255) / 256, NP), g2((per2 + 255) / 256, NP);
  swizzle_w_kernel<<<g1, 256, 0, stream>>>(pW1, pW1s, KENC, KT1, NT_H);
  swizzle_w_kernel<<<g2, 256, 0, stream>>>(pW2, pW2s, HID,  KT2, NT_H);
  swizzle_w_kernel<<<g2, 256, 0, stream>>>(pW3, pW3s, HID,  KT2, NT_H);
  swizzle_w_kernel<<<g1, 256, 0, stream>>>(vW1, vW1s, KENC, KT1, NT_H);
  swizzle_w_kernel<<<g2, 256, 0, stream>>>(vW2, vW2s, HID,  KT2, NT_H);
  swizzle_w_kernel<<<g2, 256, 0, stream>>>(vW3, vW3s, HID,  KT2, NT_H);

  dim3 gm(BATCH / MT, NP);
  mlp_kernel<<<gm, 256, 0, stream>>>(bucket, cnt, enc, pW1s, pW2s, pW3s,
                                     pb1, pb2, pb3, pWd, pbd, out, 0);
  mlp_kernel<<<gm, 256, 0, stream>>>(bucket, cnt, enc, vW1s, vW2s, vW3s,
                                     vb1, vb2, vb3, vWd, vbd, out, BATCH);
}
